// Model_72378788872203
// MI455X (gfx1250) — matrix operand order repaired and checked by reading, not compiled, not run on hardware
//
#include <hip/hip_runtime.h>

// ---------------------------------------------------------------------------
// Types for CDNA5 WMMA
// ---------------------------------------------------------------------------
typedef unsigned short bf16_t;
typedef __attribute__((ext_vector_type(8)))  unsigned short v8us;
typedef __attribute__((ext_vector_type(16))) unsigned short v16us;
typedef __attribute__((ext_vector_type(16))) __bf16        v16bf;
typedef __attribute__((ext_vector_type(8)))  float         v8f;

__device__ inline bf16_t f2bf(float f) {
  unsigned u = __float_as_uint(f);
  unsigned r = u + 0x7fffu + ((u >> 16) & 1u);   // round-to-nearest-even
  return (bf16_t)(r >> 16);
}
__device__ inline float bf2f(bf16_t h) {
  return __uint_as_float(((unsigned)h) << 16);
}

__device__ inline v16bf make_frag(v8us lo, v8us hi) {
  v16us r;
#pragma unroll
  for (int i = 0; i < 8; ++i) { r[i] = lo[i]; r[8 + i] = hi[i]; }
  return __builtin_bit_cast(v16bf, r);
}

// A fragment: 16x32 bf16, row m = lane&15; lane<16 holds K {0..15},
// lane>=16 holds K {16..31}  (two contiguous 16B chunks per lane).
// 32-bit element offsets against a uniform base -> saddr + voffset codegen.
__device__ inline v16bf load_a(const bf16_t* __restrict__ A, int lda, int m0, int k0) {
  int lane = threadIdx.x & 31;
  int off = (m0 + (lane & 15)) * lda + k0 + ((lane >> 4) << 4);
  v8us lo = *(const v8us*)(A + off);
  v8us hi = *(const v8us*)(A + off + 8);
  return make_frag(lo, hi);
}

// B fragment (normal [K,N] layout): column n = lane&15; K half = (lane>>4)*16.
__device__ inline v16bf load_b_n(const bf16_t* __restrict__ B, int ldb, int k0, int n0) {
  int lane = threadIdx.x & 31;
  int off = (k0 + ((lane >> 4) << 4)) * ldb + n0 + (lane & 15);
  v16us r;
#pragma unroll
  for (int e = 0; e < 16; ++e) r[e] = B[off + e * ldb];
  return __builtin_bit_cast(v16bf, r);
}

// B fragment from transposed storage Bt[N,K]: 16 contiguous K per lane.
__device__ inline v16bf load_b_t(const bf16_t* __restrict__ Bt, int ldb, int k0, int n0) {
  int lane = threadIdx.x & 31;
  int off = (n0 + (lane & 15)) * ldb + k0 + ((lane >> 4) << 4);
  v8us lo = *(const v8us*)(Bt + off);
  v8us hi = *(const v8us*)(Bt + off + 8);
  return make_frag(lo, hi);
}

__device__ inline v8f wmma_bf16(v16bf a, v16bf b, v8f c) {
  return __builtin_amdgcn_wmma_f32_16x16x32_bf16(false, a, false, b, (short)0, c,
                                                 false, false);
}

// C tile store: f32 C layout -> lane holds col n=lane&15, rows m0+(lane>>4)*8+r
__device__ inline void store_tile(float* __restrict__ Cf, bf16_t* __restrict__ Ch,
                                  int ldc, int m0, int n0, v8f acc, float alpha,
                                  const float* __restrict__ bias) {
  int lane = threadIdx.x & 31;
  int n = n0 + (lane & 15);
  int mb = m0 + ((lane >> 4) << 3);
  float bv = bias ? bias[n] : 0.0f;
#pragma unroll
  for (int r = 0; r < 8; ++r) {
    float v = alpha * acc[r] + bv;
    size_t idx = (size_t)(mb + r) * ldc + n;
    if (Cf) Cf[idx] = v;
    if (Ch) Ch[idx] = f2bf(v);
  }
}

// ---------------------------------------------------------------------------
// Generic batched bf16 GEMM: C[b] = alpha * A[b] @ op(B[b]) + bias
// 128 threads = 4 waves in a 2x2 grid; each wave owns a 32(M) x 16*NFRAG(N)
// tile: 2 A-frags x NFRAG B-frags -> 2*NFRAG WMMAs per 32-K step.
// NFRAG=4 when N%64==0 (all large GEMMs), else NFRAG=2.
// All M,K used by this model are multiples of 32 (S padded 720->736).
// ---------------------------------------------------------------------------
template <int TRANSB, int NFRAG>
__global__ __launch_bounds__(128) void gemm_bf16_kernel(
    const bf16_t* __restrict__ A, long long sA,
    const bf16_t* __restrict__ B, long long sB,
    const float* __restrict__ bias, float alpha,
    float* __restrict__ Cf, bf16_t* __restrict__ Ch, long long sC,
    int M, int N, int K, int lda, int ldb) {
  int wave = threadIdx.x >> 5;
  int m0 = blockIdx.y * 64 + (wave >> 1) * 32;
  int n0 = blockIdx.x * (32 * NFRAG) + (wave & 1) * (16 * NFRAG);
  if (m0 >= M || n0 >= N) return;
  const bf16_t* Ab = A + (size_t)blockIdx.z * sA;
  const bf16_t* Bb = B + (size_t)blockIdx.z * sB;

  v8f acc[2][NFRAG];
#pragma unroll
  for (int i = 0; i < 2; ++i)
#pragma unroll
    for (int j = 0; j < NFRAG; ++j) acc[i][j] = {};

  for (int k0 = 0; k0 < K; k0 += 32) {
    v16bf a0 = load_a(Ab, lda, m0, k0);
    v16bf a1 = load_a(Ab, lda, m0 + 16, k0);
    v16bf bfr[NFRAG];
#pragma unroll
    for (int j = 0; j < NFRAG; ++j)
      bfr[j] = TRANSB ? load_b_t(Bb, ldb, k0, n0 + 16 * j)
                      : load_b_n(Bb, ldb, k0, n0 + 16 * j);
#pragma unroll
    for (int j = 0; j < NFRAG; ++j) {
      acc[0][j] = wmma_bf16(a0, bfr[j], acc[0][j]);
      acc[1][j] = wmma_bf16(a1, bfr[j], acc[1][j]);
    }
  }
  float*  Cfb = Cf ? Cf + (size_t)blockIdx.z * sC : nullptr;
  bf16_t* Chb = Ch ? Ch + (size_t)blockIdx.z * sC : nullptr;
#pragma unroll
  for (int i = 0; i < 2; ++i)
#pragma unroll
    for (int j = 0; j < NFRAG; ++j)
      store_tile(Cfb, Chb, N, m0 + 16 * i, n0 + 16 * j, acc[i][j], alpha, bias);
}

// ---------------------------------------------------------------------------
// Instance norm over time: stats per (b, c)
// x [B,720,2048]; grid (C/256, B), 256 threads; thread owns one channel.
// ---------------------------------------------------------------------------
__global__ __launch_bounds__(256) void inorm_stats_kernel(
    const float* __restrict__ x, float* __restrict__ meanB, float* __restrict__ stdB) {
  int b = blockIdx.y;
  int c = blockIdx.x * 256 + threadIdx.x;
  float s = 0.f, ss = 0.f;
  const float* p = x + ((size_t)b * 720) * 2048 + c;
  for (int t = 0; t < 720; ++t) { float v = p[(size_t)t * 2048]; s += v; ss += v * v; }
  float m = s / 720.0f;
  float var = ss / 720.0f - m * m;
  meanB[(size_t)b * 2048 + c] = m;
  stdB[(size_t)b * 2048 + c] = sqrtf(var + 1e-5f);
}

// Normalize + transpose into bf16 xnT [B, 2048, 736] (K padded with zeros).
// grid (ceil(736/32)=23, 2048/32=64, B); 256 threads; LDS 32x32 tile.
__global__ __launch_bounds__(256) void inorm_transpose_kernel(
    const float* __restrict__ x, const float* __restrict__ meanB,
    const float* __restrict__ stdB, bf16_t* __restrict__ xnT) {
  __shared__ float tile[32][33];
  int s0 = blockIdx.x * 32, c0 = blockIdx.y * 32, b = blockIdx.z;
  int t = threadIdx.x;
  for (int rr = t >> 5; rr < 32; rr += 8) {
    int s = s0 + rr, c = c0 + (t & 31);
    tile[rr][t & 31] = (s < 720) ? x[((size_t)b * 720 + s) * 2048 + c] : 0.0f;
  }
  __syncthreads();
  for (int cc = t >> 5; cc < 32; cc += 8) {
    int c = c0 + cc, s = s0 + (t & 31);
    float m = meanB[(size_t)b * 2048 + c];
    float sd = stdB[(size_t)b * 2048 + c];
    float v = (s < 720) ? (tile[t & 31][cc] - m) / sd : 0.0f;
    xnT[((size_t)b * 2048 + c) * 736 + s] = f2bf(v);
  }
}

// Transpose-convert f32 [Ksrc,N] -> bf16 [N,Kpad] (K zero-padded).
// grid (Kpad/32, N/32); 256 threads; LDS 32x32 tile.
__global__ __launch_bounds__(256) void transpose_f2bf_kernel(
    const float* __restrict__ src, bf16_t* __restrict__ dst,
    int Ksrc, int N, int Kpad) {
  __shared__ float tile[32][33];
  int k0 = blockIdx.x * 32, n0 = blockIdx.y * 32, t = threadIdx.x;
  for (int rr = t >> 5; rr < 32; rr += 8) {
    int k = k0 + rr;
    tile[rr][t & 31] = (k < Ksrc) ? src[(size_t)k * N + n0 + (t & 31)] : 0.0f;
  }
  __syncthreads();
  for (int nn = t >> 5; nn < 32; nn += 8) {
    int n = n0 + nn, k = k0 + (t & 31);
    dst[(size_t)n * Kpad + k] = f2bf(tile[t & 31][nn]);
  }
}

// ---------------------------------------------------------------------------
// Softmax, in-place on bf16 rows of length L (L <= 2048); one block per row.
// ---------------------------------------------------------------------------
__global__ __launch_bounds__(256) void softmax_kernel(bf16_t* __restrict__ logits, int L) {
  __shared__ float red[256];
  int t = threadIdx.x;
  size_t base = (size_t)blockIdx.x * L;
  float vals[8];
  int cnt = 0;
  float mx = -3.4e38f;
  for (int j = t; j < L; j += 256) { float v = bf2f(logits[base + j]); vals[cnt++] = v; mx = fmaxf(mx, v); }
  red[t] = mx; __syncthreads();
  for (int o = 128; o > 0; o >>= 1) { if (t < o) red[t] = fmaxf(red[t], red[t + o]); __syncthreads(); }
  mx = red[0]; __syncthreads();
  float s = 0.f;
  for (int i = 0; i < cnt; ++i) { vals[i] = __expf(vals[i] - mx); s += vals[i]; }
  red[t] = s; __syncthreads();
  for (int o = 128; o > 0; o >>= 1) { if (t < o) red[t] += red[t + o]; __syncthreads(); }
  float inv = 1.0f / red[0];
  cnt = 0;
  for (int j = t; j < L; j += 256) logits[base + j] = f2bf(vals[cnt++] * inv);
}

// ---------------------------------------------------------------------------
// LayerNorm over D=256, optional f32 residual, dual f32/bf16 output.
// ---------------------------------------------------------------------------
__global__ __launch_bounds__(256) void ln_kernel(
    const float* __restrict__ x, const float* __restrict__ res,
    const float* __restrict__ gamma, const float* __restrict__ beta,
    float* __restrict__ outf, bf16_t* __restrict__ outh) {
  __shared__ float r1[256], r2[256];
  int t = threadIdx.x;
  size_t base = (size_t)blockIdx.x * 256;
  float v = x[base + t] + (res ? res[base + t] : 0.0f);
  r1[t] = v; r2[t] = v * v; __syncthreads();
  for (int o = 128; o > 0; o >>= 1) { if (t < o) { r1[t] += r1[t + o]; r2[t] += r2[t + o]; } __syncthreads(); }
  float m = r1[0] * (1.0f / 256.0f);
  float var = r2[0] * (1.0f / 256.0f) - m * m;
  float y = (v - m) * rsqrtf(var + 1e-5f) * gamma[t] + beta[t];
  if (outf) outf[base + t] = y;
  if (outh) outh[base + t] = f2bf(y);
}

// ---------------------------------------------------------------------------
// Covariance-loss support
// ---------------------------------------------------------------------------
// Column mean/std (ddof=1) for x [nrows, 256]; one block, thread t = column t.
__global__ __launch_bounds__(256) void colstats_kernel(
    const float* __restrict__ x, int nrows, float* __restrict__ mu, float* __restrict__ sd) {
  int t = threadIdx.x;
  float s = 0.f, ss = 0.f;
  for (int r = 0; r < nrows; ++r) { float v = x[(size_t)r * 256 + t]; s += v; ss += v * v; }
  float m = s / (float)nrows;
  float var = (ss - (float)nrows * m * m) / (float)(nrows - 1);
  mu[t] = m;
  sd[t] = sqrtf(fmaxf(var, 0.0f));
}

// xcT[d, r] = (x[r, d]-mu)/(sd+eps) as bf16; grid (nrows/32, 8); LDS transpose.
__global__ __launch_bounds__(256) void stdize_transpose_kernel(
    const float* __restrict__ x, int nrows, const float* __restrict__ mu,
    const float* __restrict__ sd, bf16_t* __restrict__ xcT) {
  __shared__ float tile[32][33];
  int r0 = blockIdx.x * 32, d0 = blockIdx.y * 32, t = threadIdx.x;
  for (int rr = t >> 5; rr < 32; rr += 8)
    tile[rr][t & 31] = x[(size_t)(r0 + rr) * 256 + d0 + (t & 31)];
  __syncthreads();
  for (int dd = t >> 5; dd < 32; dd += 8) {
    int d = d0 + dd;
    float m = mu[d], s = sd[d] + 1e-5f;
    xcT[(size_t)d * nrows + r0 + (t & 31)] = f2bf((tile[t & 31][dd] - m) / s);
  }
}

__global__ void diag_add_kernel(float* __restrict__ A, float eps) {
  int t = threadIdx.x;
  A[(size_t)t * 256 + t] += eps;
}

__global__ void setval_kernel(float* __restrict__ p, float v) { *p = v; }

// In-place Cholesky (Crout) on 256x256 SPD matrix; atomicAdd coef * logdet.
__global__ __launch_bounds__(256) void cholesky_logdet_kernel(
    float* __restrict__ A, float* __restrict__ out, float coef) {
  __shared__ float red[256];
  __shared__ float dshare;
  int t = threadIdx.x;
  float lsum = 0.0f;
  for (int j = 0; j < 256; ++j) {
    float s = 0.f;
    for (int k = t; k < j; k += 256) { float v = A[(size_t)j * 256 + k]; s += v * v; }
    red[t] = s; __syncthreads();
    for (int o = 128; o > 0; o >>= 1) { if (t < o) red[t] += red[t + o]; __syncthreads(); }
    if (t == 0) {
      float d = sqrtf(fmaxf(A[(size_t)j * 256 + j] - red[0], 1e-30f));
      A[(size_t)j * 256 + j] = d;
      dshare = d;
    }
    __syncthreads();
    float d = dshare;
    if (t == 0) lsum += logf(d);
    for (int i = j + 1 + t; i < 256; i += 256) {
      float s2 = 0.f;
      for (int k = 0; k < j; ++k) s2 += A[(size_t)i * 256 + k] * A[(size_t)j * 256 + k];
      A[(size_t)i * 256 + j] = (A[(size_t)i * 256 + j] - s2) / d;
    }
    __syncthreads();
  }
  if (t == 0) atomicAdd(out, coef * 2.0f * lsum);
}

// ---------------------------------------------------------------------------
// Elementwise helpers
// ---------------------------------------------------------------------------
__global__ void f2bf_kernel(const float* __restrict__ s, bf16_t* __restrict__ d, long long n) {
  long long i = (long long)blockIdx.x * 256 + threadIdx.x;
  if (i < n) d[i] = f2bf(s[i]);
}
__global__ void add_f2bf_kernel(const float* __restrict__ a, const float* __restrict__ b,
                                bf16_t* __restrict__ d, long long n) {
  long long i = (long long)blockIdx.x * 256 + threadIdx.x;
  if (i < n) d[i] = f2bf(a[i] + b[i]);
}

// dec_tmp [B,2048,96] -> out [B,96,2048], denormalized with per-(b,c) stats.
__global__ void finalize_kernel(const float* __restrict__ tmp, const float* __restrict__ meanB,
                                const float* __restrict__ stdB, float* __restrict__ out,
                                long long n) {
  long long i = (long long)blockIdx.x * 256 + threadIdx.x;
  if (i >= n) return;
  long long c = i % 2048;
  long long p = (i / 2048) % 96;
  long long b = i / (2048LL * 96);
  float v = tmp[((size_t)b * 2048 + c) * 96 + p];
  out[i] = v * stdB[(size_t)b * 2048 + c] + meanB[(size_t)b * 2048 + c];
}

// ---------------------------------------------------------------------------
// Host-side orchestration
// ---------------------------------------------------------------------------
static inline long long cdiv_ll(long long a, long long b) { return (a + b - 1) / b; }

static void launch_gemm(hipStream_t s, const bf16_t* A, long long sA,
                        const bf16_t* B, long long sB, int transB,
                        const float* bias, float alpha,
                        float* Cf, bf16_t* Ch, long long sC,
                        int M, int N, int K, int lda, int ldb, int batch) {
  int nf = (N % 64 == 0) ? 4 : 2;
  dim3 g((unsigned)cdiv_ll(N, 32 * nf), (unsigned)cdiv_ll(M, 64), (unsigned)batch);
  if (transB) {
    if (nf == 4)
      gemm_bf16_kernel<1, 4><<<g, 128, 0, s>>>(A, sA, B, sB, bias, alpha, Cf, Ch, sC, M, N, K, lda, ldb);
    else
      gemm_bf16_kernel<1, 2><<<g, 128, 0, s>>>(A, sA, B, sB, bias, alpha, Cf, Ch, sC, M, N, K, lda, ldb);
  } else {
    if (nf == 4)
      gemm_bf16_kernel<0, 4><<<g, 128, 0, s>>>(A, sA, B, sB, bias, alpha, Cf, Ch, sC, M, N, K, lda, ldb);
    else
      gemm_bf16_kernel<0, 2><<<g, 128, 0, s>>>(A, sA, B, sB, bias, alpha, Cf, Ch, sC, M, N, K, lda, ldb);
  }
}

extern "C" void kernel_launch(void* const* d_in, const int* in_sizes, int n_in,
                              void* d_out, int out_size, void* d_ws, size_t ws_size,
                              hipStream_t stream) {
  (void)in_sizes; (void)n_in; (void)out_size; (void)ws_size;
  const int Bb = 32, D = 256, Pp = 96, Cc = 2048;
  const float* x_enc  = (const float*)d_in[0];
  const float* in_w   = (const float*)d_in[4];
  const float* in_b   = (const float*)d_in[5];
  const float* lq[3]  = {(const float*)d_in[6], (const float*)d_in[7], (const float*)d_in[8]};
  const float* down_w = (const float*)d_in[9];
  const float* down_b = (const float*)d_in[10];
  const float* down_ln= (const float*)d_in[11];
  const float* pred_w = (const float*)d_in[12];
  const float* pred_b = (const float*)d_in[13];
  const float* up_w   = (const float*)d_in[14];
  const float* up_b   = (const float*)d_in[15];
  const float* up_ln  = (const float*)d_in[16];
  const float* out_w  = (const float*)d_in[17];
  const float* out_b  = (const float*)d_in[18];
  float* out = (float*)d_out;
  float* cov_out = out + (long long)Bb * Pp * Cc;   // scalar slot

  const int LqD[3] = {512, 128, 32};                // down latent lengths
  const int LqU[3] = {128, 512, 2048};              // up query lengths

  size_t cur = 0;
  auto alloc = [&](size_t bytes) -> void* {
    cur = (cur + 255) & ~(size_t)255;
    void* p = (char*)d_ws + cur;
    cur += bytes;
    return p;
  };

  // -------- persistent allocations --------
  float*  meanB = (float*)alloc((size_t)Bb * Cc * 4);
  float*  stdB  = (float*)alloc((size_t)Bb * Cc * 4);
  bf16_t* w_in_bf   = (bf16_t*)alloc((size_t)D * 736 * 2);        // [256,736] (N,K)
  bf16_t* w_down_bf = (bf16_t*)alloc((size_t)3 * 4 * D * D * 2);  // transposed
  bf16_t* w_up_bf   = (bf16_t*)alloc((size_t)3 * 4 * D * D * 2);  // transposed
  bf16_t* w_pred_bf = (bf16_t*)alloc((size_t)D * D * 2);          // transposed
  bf16_t* w_out_bf  = (bf16_t*)alloc((size_t)Pp * D * 2);         // [96,256] (N,K)
  bf16_t* lq_bf[3];
  for (int i = 0; i < 3; ++i) lq_bf[i] = (bf16_t*)alloc((size_t)LqD[i] * D * 2);
  float*  h_f  = (float*)alloc((size_t)Bb * Cc * D * 4);
  bf16_t* h_bf = (bf16_t*)alloc((size_t)Bb * Cc * D * 2);
  float*  s_f[3]; bf16_t* s_bf[3];
  for (int i = 0; i < 3; ++i) {
    s_f[i]  = (float*)alloc((size_t)Bb * LqD[i] * D * 4);
    s_bf[i] = (bf16_t*)alloc((size_t)Bb * LqD[i] * D * 2);
  }
  bf16_t* xpred_bf = (bf16_t*)alloc((size_t)Bb * 32 * D * 2);
  float*  xup_f[3]; bf16_t* xup_bf[2];
  for (int i = 0; i < 3; ++i) xup_f[i] = (float*)alloc((size_t)Bb * LqU[i] * D * 4);
  for (int i = 0; i < 2; ++i) xup_bf[i] = (bf16_t*)alloc((size_t)Bb * LqU[i] * D * 2);
  size_t arena = cur;

  // -------- weight conversions (transposed to [N,K] bf16) --------
  transpose_f2bf_kernel<<<dim3(736 / 32, D / 32), 256, 0, stream>>>(in_w, w_in_bf, 720, D, 736);
  for (int i = 0; i < 12; ++i) {
    transpose_f2bf_kernel<<<dim3(D / 32, D / 32), 256, 0, stream>>>(
        down_w + (size_t)i * D * D, w_down_bf + (size_t)i * D * D, D, D, D);
    transpose_f2bf_kernel<<<dim3(D / 32, D / 32), 256, 0, stream>>>(
        up_w + (size_t)i * D * D, w_up_bf + (size_t)i * D * D, D, D, D);
  }
  transpose_f2bf_kernel<<<dim3(D / 32, D / 32), 256, 0, stream>>>(pred_w, w_pred_bf, D, D, D);
  transpose_f2bf_kernel<<<dim3(D / 32, Pp / 32), 256, 0, stream>>>(out_w, w_out_bf, D, Pp, D);
  for (int i = 0; i < 3; ++i) {
    long long n = (long long)LqD[i] * D;
    f2bf_kernel<<<(unsigned)cdiv_ll(n, 256), 256, 0, stream>>>(lq[i], lq_bf[i], n);
  }

  // -------- instance norm + input projection h = xn^T @ in_w + b --------
  inorm_stats_kernel<<<dim3(Cc / 256, Bb), 256, 0, stream>>>(x_enc, meanB, stdB);
  {
    cur = arena;
    bf16_t* xnT = (bf16_t*)alloc((size_t)Bb * Cc * 736 * 2);
    inorm_transpose_kernel<<<dim3(23, Cc / 32, Bb), 256, 0, stream>>>(x_enc, meanB, stdB, xnT);
    launch_gemm(stream, xnT, (long long)Cc * 736, w_in_bf, 0, 1, in_b, 1.0f,
                h_f, h_bf, (long long)Cc * D, Cc, D, 736, 736, 736, Bb);
  }

  // -------- down (latent-query) attention layers --------
  const bf16_t* x_bf = h_bf;
  int Lin = Cc;
  for (int i = 0; i < 3; ++i) {
    int Lq = LqD[i];
    cur = arena;
    bf16_t* q_bf  = (bf16_t*)alloc((size_t)Lq * D * 2);
    bf16_t* k_bf  = (bf16_t*)alloc((size_t)Bb * Lin * D * 2);
    bf16_t* v_bf  = (bf16_t*)alloc((size_t)Bb * Lin * D * 2);
    bf16_t* logit = (bf16_t*)alloc((size_t)Bb * Lq * Lin * 2);
    bf16_t* av    = (bf16_t*)alloc((size_t)Bb * Lq * D * 2);
    float*  o     = (float*)alloc((size_t)Bb * Lq * D * 4);
    const bf16_t* Wq = w_down_bf + (size_t)(i * 4 + 0) * D * D;
    const bf16_t* Wk = w_down_bf + (size_t)(i * 4 + 1) * D * D;
    const bf16_t* Wv = w_down_bf + (size_t)(i * 4 + 2) * D * D;
    const bf16_t* Wo = w_down_bf + (size_t)(i * 4 + 3) * D * D;
    const float* bq = down_b + (size_t)(i * 4 + 0) * D;
    const float* bk = down_b + (size_t)(i * 4 + 1) * D;
    const float* bv = down_b + (size_t)(i * 4 + 2) * D;
    const float* bo = down_b + (size_t)(i * 4 + 3) * D;

    launch_gemm(stream, lq_bf[i], 0, Wq, 0, 1, bq, 1.0f, nullptr, q_bf, 0,
                Lq, D, D, D, D, 1);
    launch_gemm(stream, x_bf, (long long)Lin * D, Wk, 0, 1, bk, 1.0f,
                nullptr, k_bf, (long long)Lin * D, Lin, D, D, D, D, Bb);
    launch_gemm(stream, x_bf, (long long)Lin * D, Wv, 0, 1, bv, 1.0f,
                nullptr, v_bf, (long long)Lin * D, Lin, D, D, D, D, Bb);
    // logits = q @ k^T / sqrt(D)
    launch_gemm(stream, q_bf, 0, k_bf, (long long)Lin * D, 1, nullptr, 0.0625f,
                nullptr, logit, (long long)Lq * Lin, Lq, Lin, D, D, D, Bb);
    softmax_kernel<<<Bb * Lq, 256, 0, stream>>>(logit, Lin);
    launch_gemm(stream, logit, (long long)Lq * Lin, v_bf, (long long)Lin * D, 0,
                nullptr, 1.0f, nullptr, av, (long long)Lq * D, Lq, D, Lin, Lin, D, Bb);
    launch_gemm(stream, av, (long long)Lq * D, Wo, 0, 1, bo, 1.0f,
                o, nullptr, (long long)Lq * D, Lq, D, D, D, D, Bb);
    ln_kernel<<<Bb * Lq, 256, 0, stream>>>(o, nullptr,
                                           down_ln + (size_t)(i * 2 + 0) * D,
                                           down_ln + (size_t)(i * 2 + 1) * D,
                                           s_f[i], s_bf[i]);
    x_bf = s_bf[i];
    Lin = Lq;
  }

  // -------- covariance (slogdet) loss over the 3 skips --------
  setval_kernel<<<1, 1, 0, stream>>>(cov_out, 0.0f);
  for (int i = 0; i < 3; ++i) {
    int n = Bb * LqD[i];
    cur = arena;
    float*  mu  = (float*)alloc(D * 4);
    float*  sdv = (float*)alloc(D * 4);
    bf16_t* xcT = (bf16_t*)alloc((size_t)D * n * 2);
    float*  cov = (float*)alloc((size_t)D * D * 4);
    colstats_kernel<<<1, 256, 0, stream>>>(s_f[i], n, mu, sdv);
    stdize_transpose_kernel<<<dim3(n / 32, D / 32), 256, 0, stream>>>(s_f[i], n, mu, sdv, xcT);
    // cov = xcT @ xcT^T / (n-1)   (A = B = xcT with transB)
    launch_gemm(stream, xcT, 0, xcT, 0, 1, nullptr, 1.0f / (float)(n - 1),
                cov, nullptr, 0, D, D, n, n, n, 1);
    diag_add_kernel<<<1, 256, 0, stream>>>(cov, 1e-5f);
    cholesky_logdet_kernel<<<1, 256, 0, stream>>>(cov, cov_out, -0.1f / (3.0f * (float)D));
  }

  // -------- bottom prediction: x = s3 @ pred_w + pred_b --------
  launch_gemm(stream, s_bf[2], 32LL * D, w_pred_bf, 0, 1, pred_b, 1.0f,
              nullptr, xpred_bf, 32LL * D, 32, D, D, D, D, Bb);

  // -------- up (cross) attention layers --------
  const bf16_t* xk_bf = xpred_bf;
  int Lk = 32;
  const bf16_t* qin_bf_arr[3] = {s_bf[1], s_bf[0], h_bf};
  const float*  qin_f_arr[3]  = {s_f[1],  s_f[0],  h_f};
  for (int i = 0; i < 3; ++i) {
    int Lq = LqU[i];
    cur = arena;
    bf16_t* q_bf  = (bf16_t*)alloc((size_t)Bb * Lq * D * 2);
    bf16_t* k_bf  = (bf16_t*)alloc((size_t)Bb * Lk * D * 2);
    bf16_t* v_bf  = (bf16_t*)alloc((size_t)Bb * Lk * D * 2);
    bf16_t* logit = (bf16_t*)alloc((size_t)Bb * Lq * Lk * 2);
    bf16_t* av    = (bf16_t*)alloc((size_t)Bb * Lq * D * 2);
    float*  o     = (float*)alloc((size_t)Bb * Lq * D * 4);
    const bf16_t* Wq = w_up_bf + (size_t)(i * 4 + 0) * D * D;
    const bf16_t* Wk = w_up_bf + (size_t)(i * 4 + 1) * D * D;
    const bf16_t* Wv = w_up_bf + (size_t)(i * 4 + 2) * D * D;
    const bf16_t* Wo = w_up_bf + (size_t)(i * 4 + 3) * D * D;
    const float* bq = up_b + (size_t)(i * 4 + 0) * D;
    const float* bk = up_b + (size_t)(i * 4 + 1) * D;
    const float* bv = up_b + (size_t)(i * 4 + 2) * D;
    const float* bo = up_b + (size_t)(i * 4 + 3) * D;

    launch_gemm(stream, qin_bf_arr[i], (long long)Lq * D, Wq, 0, 1, bq, 1.0f,
                nullptr, q_bf, (long long)Lq * D, Lq, D, D, D, D, Bb);
    launch_gemm(stream, xk_bf, (long long)Lk * D, Wk, 0, 1, bk, 1.0f,
                nullptr, k_bf, (long long)Lk * D, Lk, D, D, D, D, Bb);
    launch_gemm(stream, xk_bf, (long long)Lk * D, Wv, 0, 1, bv, 1.0f,
                nullptr, v_bf, (long long)Lk * D, Lk, D, D, D, D, Bb);
    launch_gemm(stream, q_bf, (long long)Lq * D, k_bf, (long long)Lk * D, 1,
                nullptr, 0.0625f, nullptr, logit, (long long)Lq * Lk,
                Lq, Lk, D, D, D, Bb);
    softmax_kernel<<<Bb * Lq, 256, 0, stream>>>(logit, Lk);
    launch_gemm(stream, logit, (long long)Lq * Lk, v_bf, (long long)Lk * D, 0,
                nullptr, 1.0f, nullptr, av, (long long)Lq * D, Lq, D, Lk, Lk, D, Bb);
    launch_gemm(stream, av, (long long)Lq * D, Wo, 0, 1, bo, 1.0f,
                o, nullptr, (long long)Lq * D, Lq, D, D, D, D, Bb);
    ln_kernel<<<Bb * Lq, 256, 0, stream>>>(o, qin_f_arr[i],
                                           up_ln + (size_t)(i * 2 + 0) * D,
                                           up_ln + (size_t)(i * 2 + 1) * D,
                                           xup_f[i], (i < 2) ? xup_bf[i] : nullptr);
    if (i < 2) xk_bf = xup_bf[i];
    Lk = Lq;
  }

  // -------- output projection + denormalize + transpose --------
  {
    cur = arena;
    bf16_t* xh   = (bf16_t*)alloc((size_t)Bb * Cc * D * 2);
    float*  dtmp = (float*)alloc((size_t)Bb * Cc * Pp * 4);
    long long n = (long long)Bb * Cc * D;
    add_f2bf_kernel<<<(unsigned)cdiv_ll(n, 256), 256, 0, stream>>>(xup_f[2], h_f, xh, n);
    launch_gemm(stream, xh, (long long)Cc * D, w_out_bf, 0, 1, out_b, 1.0f,
                dtmp, nullptr, (long long)Cc * Pp, Cc, Pp, D, D, D, Bb);
    long long no = (long long)Bb * Pp * Cc;
    finalize_kernel<<<(unsigned)cdiv_ll(no, 256), 256, 0, stream>>>(dtmp, meanB, stdB, out, no);
  }
}
